// MultiHeadAttention_22247930593412
// MI455X (gfx1250) — matrix operand order repaired and checked by reading, not compiled, not run on hardware
//
#include <hip/hip_runtime.h>

typedef __attribute__((ext_vector_type(16))) __bf16 v16bf;
typedef __attribute__((ext_vector_type(8)))  __bf16 v8bf;
typedef __attribute__((ext_vector_type(4)))  __bf16 v4bf;
typedef __attribute__((ext_vector_type(8)))  float  v8f;

#define N_EMBD 1024
#define N_HEAD 16
#define HEADD  64
#define SEQ    2048
#define BATCH  4
#define ROWS   (BATCH * SEQ)   // 8192

// ---------------------------------------------------------------------------
// WMMA helpers (CDNA5: V_WMMA_F32_16X16X32_BF16, wave32)
// ---------------------------------------------------------------------------
__device__ __forceinline__ v8f wmma_bf16(v16bf a, v16bf b, v8f c) {
  // (neg_a, A, neg_b, B, c_mod, C, reuse_a, reuse_b)
  return __builtin_amdgcn_wmma_f32_16x16x32_bf16(false, a, false, b, (short)0, c,
                                                 false, false);
}

// A-operand (16 rows x 32 K) from row-major bf16, leading dim `ld` (elements).
// A layout: lane(half,row): elems 0..15 = src[row][16*half + 0..15] (contig)
__device__ __forceinline__ v16bf load_a32(const __bf16* base, int ld, int half, int row) {
  const __bf16* rp = base + row * ld + 16 * half;
  v8bf lo = *(const v8bf*)rp;
  v8bf hi = *(const v8bf*)(rp + 8);
  v16bf r;
#pragma unroll
  for (int i = 0; i < 8; ++i) { r[i] = lo[i]; r[i + 8] = hi[i]; }
  return r;
}

// B-operand (32 K x 16 N) supplied as Bt (row = output column n), row-major.
// B order used here (same as A): lanes 0-15 hold K=0..15, lanes 16-31 hold K=16..31 (contig).
__device__ __forceinline__ v16bf load_b32(const __bf16* baseT, int ld, int half, int n) {
  const __bf16* rp = baseT + n * ld + 16 * half;
  v8bf lo = *(const v8bf*)rp;
  v8bf hi = *(const v8bf*)(rp + 8);
  v16bf r;
#pragma unroll
  for (int i = 0; i < 8; ++i) { r[i] = lo[i]; r[i + 8] = hi[i]; }
  return r;
}

// ---------------------------------------------------------------------------
// fp32 -> bf16 bulk convert
// ---------------------------------------------------------------------------
__global__ __launch_bounds__(256) void cvt_f32_to_bf16(const float* __restrict__ x,
                                                       __bf16* __restrict__ y, int n4) {
  int i = blockIdx.x * blockDim.x + threadIdx.x;
  if (i < n4) {
    const float4 v = ((const float4*)x)[i];
    v4bf o;
    o[0] = (__bf16)v.x; o[1] = (__bf16)v.y; o[2] = (__bf16)v.z; o[3] = (__bf16)v.w;
    ((v4bf*)y)[i] = o;
  }
}

// ---------------------------------------------------------------------------
// W (1024x1024 f32, [k][n]) -> Wt (bf16, [n][k]) via LDS tile transpose
// ---------------------------------------------------------------------------
__global__ __launch_bounds__(256) void transpose_w(const float* __restrict__ W,
                                                   __bf16* __restrict__ Wt) {
  __shared__ float tile[32][33];
  const int tx = threadIdx.x & 31, ty = threadIdx.x >> 5;  // 32x8
  const int k0 = blockIdx.y * 32, n0 = blockIdx.x * 32;
#pragma unroll
  for (int i = 0; i < 32; i += 8)
    tile[ty + i][tx] = W[(k0 + ty + i) * N_EMBD + n0 + tx];
  __syncthreads();
#pragma unroll
  for (int i = 0; i < 32; i += 8)
    Wt[(n0 + ty + i) * N_EMBD + k0 + tx] = (__bf16)tile[tx][ty + i];
}

// ---------------------------------------------------------------------------
// GEMM building blocks: one k-chunk = A 32x32 (2 tiles) + B 32x64 (4 tiles)
// ---------------------------------------------------------------------------
struct Chunk { v16bf a0, a1, b[4]; };

__device__ __forceinline__ void load_chunk(Chunk& c, const __bf16* Ap, const __bf16* Bp,
                                           int k, int half, int mr) {
  c.a0 = load_a32(Ap + k, N_EMBD, half, mr);
  c.a1 = load_a32(Ap + 16 * N_EMBD + k, N_EMBD, half, mr);
#pragma unroll
  for (int j = 0; j < 4; ++j)
    c.b[j] = load_b32(Bp + 16 * j * N_EMBD + k, N_EMBD, half, mr);
}

__device__ __forceinline__ void mma_chunk(v8f acc[2][4], const Chunk& c) {
#pragma unroll
  for (int j = 0; j < 4; ++j) acc[0][j] = wmma_bf16(c.a0, c.b[j], acc[0][j]);
#pragma unroll
  for (int j = 0; j < 4; ++j) acc[1][j] = wmma_bf16(c.a1, c.b[j], acc[1][j]);
}

// ---------------------------------------------------------------------------
// C = A(bf16, Mx1024) @ Bt(bf16 pre-transposed 1024x1024)   [K = 1024]
// 4 waves/block; wave computes a 32x64 tile (2x4 accumulators).
// Ping-pong double buffer: consume chunk -> immediately reload same buffer,
// so loads stay one full chunk (6 b128-pairs) ahead of the 8-WMMA bursts
// with no register-rotation copies.
// mode 0: f32 out[m][n] = acc + bias[n]
// mode 1: bf16 out (b,h,s,d)     (Q / K layout)
// mode 2: bf16 out (b,h,d,s)     (V transposed layout)
// ---------------------------------------------------------------------------
__global__ __launch_bounds__(128) void gemm_bf16(const __bf16* __restrict__ A,
                                                 const __bf16* __restrict__ Bt,
                                                 void* __restrict__ Out,
                                                 const float* __restrict__ bias,
                                                 int mode) {
  const int lane = threadIdx.x & 31, wave = threadIdx.x >> 5;
  const int half = lane >> 4, mr = lane & 15;
  const int m0 = blockIdx.y * 128 + wave * 32;   // 32 rows per wave
  const int n0 = blockIdx.x * 64;

  const __bf16* Ap = A + m0 * N_EMBD;
  const __bf16* Bp = Bt + n0 * N_EMBD;

  v8f acc[2][4] = {};
  Chunk c0, c1;
  load_chunk(c0, Ap, Bp, 0, half, mr);
  load_chunk(c1, Ap, Bp, 32, half, mr);

#pragma unroll 1
  for (int k = 64; k < N_EMBD; k += 64) {
    mma_chunk(acc, c0);
    load_chunk(c0, Ap, Bp, k, half, mr);
    mma_chunk(acc, c1);
    load_chunk(c1, Ap, Bp, k + 32, half, mr);
  }
  mma_chunk(acc, c0);
  mma_chunk(acc, c1);

#pragma unroll
  for (int mi = 0; mi < 2; ++mi) {
#pragma unroll
    for (int j = 0; j < 4; ++j) {
      const int n = n0 + 16 * j + mr;          // C layout: lane holds column n
#pragma unroll
      for (int r = 0; r < 8; ++r) {
        const int m = m0 + 16 * mi + r + 8 * half;  // VGPR r -> row r + 8*half
        float v = acc[mi][j][r];
        if (mode == 0) {
          ((float*)Out)[m * N_EMBD + n] = v + bias[n];
        } else {
          const int b = m >> 11, s = m & 2047;
          const int h = n >> 6, d = n & 63;
          const int bh = b * N_HEAD + h;
          if (mode == 1)
            ((__bf16*)Out)[(bh * SEQ + s) * HEADD + d] = (__bf16)v;
          else
            ((__bf16*)Out)[(bh * HEADD + d) * SEQ + s] = (__bf16)v;
        }
      }
    }
  }
}

// ---------------------------------------------------------------------------
// Causal flash attention: block = 4 waves, wave owns 16 query rows.
// Computes S^T = K_tile @ Q^T so each lane-pair holds one query's scores
// (softmax row = per-lane + one shfl_xor(16)), accumulates O^T = V^T @ P^T.
// K tiles for the next iteration and V^T tiles for this one are issued
// before the softmax VALU block, which hides their latency.
// ---------------------------------------------------------------------------
__global__ __launch_bounds__(128) void attn_fwd(const __bf16* __restrict__ Q,   // (B,H,S,D)
                                                const __bf16* __restrict__ K,   // (B,H,S,D)
                                                const __bf16* __restrict__ Vt,  // (B,H,D,S)
                                                __bf16* __restrict__ O) {       // (B,S,E)
  const int lane = threadIdx.x & 31, wave = threadIdx.x >> 5;
  const int half = lane >> 4, mr = lane & 15;
  const int h = blockIdx.y, b = blockIdx.z;
  const int qbase = blockIdx.x * 64 + wave * 16;
  const int bh = b * N_HEAD + h;
  const __bf16* Qp = Q + (bh * SEQ + qbase) * HEADD;
  const __bf16* Kp = K + bh * SEQ * HEADD;
  const __bf16* Vp = Vt + bh * HEADD * SEQ;

  // Q tile as B operand (Bt = Q row-major), resident for the whole key loop
  const v16bf qb0 = load_b32(Qp, HEADD, half, mr);        // d = 0..31
  const v16bf qb1 = load_b32(Qp + 32, HEADD, half, mr);   // d = 32..63

  v8f o[4] = {};                      // O^T tiles: d-chunks 0..3, col = query
  float m_i = -1e30f, l_i = 0.f;      // per-query stats (lane pair duplicates)
  const int qabs = qbase + mr;
  const int kend = qbase + 15;        // causal: last key this wave needs
  const float sc = 0.125f;            // 1/sqrt(64)

  // K tiles for kt = 0 (4 A-operand tiles: {keys 0-15, 16-31} x {d 0-31, 32-63})
  v16bf ka[4];
#pragma unroll
  for (int t = 0; t < 4; ++t)
    ka[t] = load_a32(Kp + (t >> 1) * 16 * HEADD + (t & 1) * 32, HEADD, half, mr);

#pragma unroll 1
  for (int kt = 0; kt <= kend; kt += 32) {
    // S^T tiles (16 keys x 16 queries): st0 keys kt..kt+15, st1 keys kt+16..31
    v8f st0 = {}, st1 = {};
    st0 = wmma_bf16(ka[0], qb0, st0);
    st0 = wmma_bf16(ka[1], qb1, st0);
    st1 = wmma_bf16(ka[2], qb0, st1);
    st1 = wmma_bf16(ka[3], qb1, st1);

    // issue V^T tile loads + next iteration's K tile loads now;
    // the softmax VALU below hides their latency
    v16bf av[4];
#pragma unroll
    for (int t = 0; t < 4; ++t)
      av[t] = load_a32(Vp + (16 * t) * SEQ + kt, SEQ, half, mr);

    const int ktn = (kt + 32 <= kend) ? kt + 32 : kt;  // clamp (no branch)
#pragma unroll
    for (int t = 0; t < 4; ++t)
      ka[t] = load_a32(Kp + (ktn + (t >> 1) * 16) * HEADD + (t & 1) * 32,
                       HEADD, half, mr);

    // causal mask + scale + online softmax (row lives in lane-pair L, L^16)
    float v0[8], v1[8], pm = -1e30f;
#pragma unroll
    for (int r = 0; r < 8; ++r) {
      const int key0 = kt + r + 8 * half;   // C layout: VGPR r -> row r + 8*half
      const int key1 = key0 + 16;
      v0[r] = (key0 <= qabs) ? st0[r] * sc : -1e30f;
      v1[r] = (key1 <= qabs) ? st1[r] * sc : -1e30f;
      pm = fmaxf(pm, fmaxf(v0[r], v1[r]));
    }
    pm = fmaxf(pm, __shfl_xor(pm, 16, 32));
    const float m_new = fmaxf(m_i, pm);
    const float corr = __expf(m_i - m_new);
    float lsum = 0.f;
    float e0[8], e1[8];
#pragma unroll
    for (int r = 0; r < 8; ++r) {
      e0[r] = __expf(v0[r] - m_new);
      e1[r] = __expf(v1[r] - m_new);
      lsum += e0[r] + e1[r];
    }
    lsum += __shfl_xor(lsum, 16, 32);
    l_i = l_i * corr + lsum;
    m_i = m_new;

    // Rebuild P^T as a B operand: lane needs P[q][16*half + 0..15].
    // half0 owns keys 0..7 (e0) / 16..23 (e1); half1 owns 8..15 / 24..31.
    v16bf pb;
#pragma unroll
    for (int r = 0; r < 8; ++r) {
      const float send = half ? e0[r] : e1[r];
      const float rec = __shfl_xor(send, 16, 32);
      pb[r]     = (__bf16)(half ? rec : e0[r]);   // key 16*half + r
      pb[r + 8] = (__bf16)(half ? e1[r] : rec);   // key 16*half + 8 + r
    }

    // O^T += V^T_chunk @ P^T   (A = Vt rows d, K-dim = 32 keys)
#pragma unroll
    for (int t = 0; t < 4; ++t) {
#pragma unroll
      for (int r = 0; r < 8; ++r) o[t][r] *= corr;
      o[t] = wmma_bf16(av[t], pb, o[t]);
    }
  }

  // normalize and store bf16 to (B,S,E); 8 contiguous bf16 per tile = one b128
  const float inv = 1.f / l_i;
  const int s = qbase + mr;
#pragma unroll
  for (int t = 0; t < 4; ++t) {
    v8bf ov;
#pragma unroll
    for (int r = 0; r < 8; ++r) ov[r] = (__bf16)(o[t][r] * inv);
    *(v8bf*)(O + (b * SEQ + s) * N_EMBD + h * HEADD + 16 * t + 8 * half) = ov;
  }
}

// ---------------------------------------------------------------------------
extern "C" void kernel_launch(void* const* d_in, const int* in_sizes, int n_in,
                              void* d_out, int out_size, void* d_ws, size_t ws_size,
                              hipStream_t stream) {
  (void)in_sizes; (void)n_in; (void)out_size; (void)ws_size;
  const float* x  = (const float*)d_in[0];
  const float* Wq = (const float*)d_in[1];
  const float* Wk = (const float*)d_in[2];
  const float* Wv = (const float*)d_in[3];
  const float* Wo = (const float*)d_in[4];
  const float* bo = (const float*)d_in[5];

  char* ws = (char*)d_ws;
  const size_t SZ_X = (size_t)ROWS * N_EMBD * 2;    // 16 MiB
  const size_t SZ_W = (size_t)N_EMBD * N_EMBD * 2;  //  2 MiB
  size_t off = 0;
  __bf16* Xbf = (__bf16*)(ws + off); off += SZ_X;
  __bf16* Wqt = (__bf16*)(ws + off); off += SZ_W;
  __bf16* Wkt = (__bf16*)(ws + off); off += SZ_W;
  __bf16* Wvt = (__bf16*)(ws + off); off += SZ_W;
  __bf16* Wot = (__bf16*)(ws + off); off += SZ_W;
  __bf16* Qb  = (__bf16*)(ws + off); off += SZ_X;
  __bf16* Kb  = (__bf16*)(ws + off); off += SZ_X;
  __bf16* Vtb = (__bf16*)(ws + off); off += SZ_X;
  __bf16* Ob  = (__bf16*)(ws + off); off += SZ_X;   // total ~88 MiB

  const int n4 = ROWS * N_EMBD / 4;
  cvt_f32_to_bf16<<<(n4 + 255) / 256, 256, 0, stream>>>(x, Xbf, n4);

  dim3 tg(N_EMBD / 32, N_EMBD / 32);
  transpose_w<<<tg, 256, 0, stream>>>(Wq, Wqt);
  transpose_w<<<tg, 256, 0, stream>>>(Wk, Wkt);
  transpose_w<<<tg, 256, 0, stream>>>(Wv, Wvt);
  transpose_w<<<tg, 256, 0, stream>>>(Wo, Wot);

  dim3 gg(N_EMBD / 64, ROWS / 128);
  gemm_bf16<<<gg, 128, 0, stream>>>(Xbf, Wqt, Qb, nullptr, 1);
  gemm_bf16<<<gg, 128, 0, stream>>>(Xbf, Wkt, Kb, nullptr, 1);
  gemm_bf16<<<gg, 128, 0, stream>>>(Xbf, Wvt, Vtb, nullptr, 2);

  attn_fwd<<<dim3(SEQ / 64, N_HEAD, BATCH), 128, 0, stream>>>(Qb, Kb, Vtb, Ob);

  gemm_bf16<<<gg, 128, 0, stream>>>(Ob, Wot, d_out, bo, 0);
}
